// Block_19756849561899
// MI455X (gfx1250) — matrix operand order repaired and checked by reading, not compiled, not run on hardware
//
#include <hip/hip_runtime.h>
#include <hip/hip_bf16.h>
#include <math.h>

typedef __attribute__((ext_vector_type(16))) _Float16 v16h;
typedef __attribute__((ext_vector_type(8)))  _Float16 v8h;
typedef __attribute__((ext_vector_type(8)))  float    v8f;
typedef __attribute__((ext_vector_type(4)))  int      v4i;

#define DIMC  384
#define HEADS 6
#define DH    64
#define HID   1536
#define CQKV  1152
#define BATCH 4
#define SEQ   2048
#define MTOT  (BATCH*SEQ)

#define AS1 __attribute__((address_space(1)))
#define AS3 __attribute__((address_space(3)))

// ---------------------------------------------------------------------------
// CDNA5 async global->LDS copy (ASYNCcnt-tracked), 16B per lane
// builtin signature: (int4 AS1* gsrc, int4 AS3* lds_dst, imm offset, imm cpol)
// ---------------------------------------------------------------------------
__device__ __forceinline__ void async_copy16(void* lds_dst, const void* gsrc) {
#if __has_builtin(__builtin_amdgcn_global_load_async_to_lds_b128)
  __builtin_amdgcn_global_load_async_to_lds_b128(
      (AS1 v4i*)(AS1 void*)(void*)gsrc,
      (AS3 v4i*)(AS3 void*)lds_dst, 0, 0);
#else
  unsigned l = (unsigned)(unsigned long long)(AS3 void*)lds_dst;
  asm volatile("global_load_async_to_lds_b128 %0, %1, off"
               :: "v"(l), "v"(gsrc) : "memory");
#endif
}

__device__ __forceinline__ void wait_async0() {
#if __has_builtin(__builtin_amdgcn_s_wait_asynccnt)
  __builtin_amdgcn_s_wait_asynccnt(0);
#else
  asm volatile("s_wait_asynccnt 0" ::: "memory");
#endif
}

// ---------------------------------------------------------------------------
// WMMA helpers (CDNA5 wave32, 16x16x32 f16 -> f32)
// ---------------------------------------------------------------------------
__device__ __forceinline__ v8f wmma16(v16h a, v16h b, v8f c) {
  return __builtin_amdgcn_wmma_f32_16x16x32_f16(false, a, false, b, (short)0, c,
                                                false, false);
}

// A-fragment, this code's contiguous k order (B matches): lane L (row M=L%16) holds
//   lanes 0-15 : K = 0..15 ; lanes 16-31: K = 16..31
__device__ __forceinline__ v16h load_a_pair(const _Float16* p0, const _Float16* p1) {
  union { v16h v; v8h h[2]; } u;
  u.h[0] = *(const v8h*)p0;
  u.h[1] = *(const v8h*)p1;
  return u.v;
}

// ---------------------------------------------------------------------------
// Weight prep: fp32 [K,N] -> fp16 transposed [N,K]
// ---------------------------------------------------------------------------
__global__ void cvt_t_kernel(const float* __restrict__ src, _Float16* __restrict__ dst,
                             int K, int N) {
  int idx = blockIdx.x * 256 + threadIdx.x;
  if (idx < K * N) {
    int k = idx / N, n = idx - k * N;
    dst[(size_t)n * K + k] = (_Float16)src[idx];
  }
}

// ---------------------------------------------------------------------------
// LayerNorm: one block (128 threads) per row of 384, fp32 in -> fp16 out
// ---------------------------------------------------------------------------
__global__ __launch_bounds__(128) void ln_kernel(const float* __restrict__ x,
                                                 const float* __restrict__ g,
                                                 const float* __restrict__ b,
                                                 _Float16* __restrict__ out) {
  int row = blockIdx.x;
  int t = threadIdx.x;
  const float* xr = x + (size_t)row * DIMC;
  float v0 = xr[t], v1 = xr[t + 128], v2 = xr[t + 256];
  float s  = v0 + v1 + v2;
  float ss = v0 * v0 + v1 * v1 + v2 * v2;
#pragma unroll
  for (int m = 1; m < 32; m <<= 1) {
    s  += __shfl_xor(s,  m, 32);
    ss += __shfl_xor(ss, m, 32);
  }
  __shared__ float as_[4], asq_[4];
  int w = t >> 5;
  if ((t & 31) == 0) { as_[w] = s; asq_[w] = ss; }
  __syncthreads();
  s  = as_[0] + as_[1] + as_[2] + as_[3];
  ss = asq_[0] + asq_[1] + asq_[2] + asq_[3];
  float mu   = s * (1.0f / DIMC);
  float var  = ss * (1.0f / DIMC) - mu * mu;
  float rstd = rsqrtf(var + 1e-6f);
  _Float16* o = out + (size_t)row * DIMC;
  o[t]       = (_Float16)((v0 - mu) * rstd * g[t]       + b[t]);
  o[t + 128] = (_Float16)((v1 - mu) * rstd * g[t + 128] + b[t + 128]);
  o[t + 256] = (_Float16)((v2 - mu) * rstd * g[t + 256] + b[t + 256]);
}

// ---------------------------------------------------------------------------
// Tiled WMMA GEMM: C[M,N] = A[M,K](f16) * BT[N,K](f16, pre-transposed) + bias
// 128 threads = 4 waves; block tile 128x128; wave tile 64x64 (4x4 WMMA accums)
// Double-buffered LDS tiles, async global->LDS staging, K-step 32.
// MODE 0: f16 = r+bias   MODE 1: f16 = gelu(r+bias)   MODE 2: f32 = res+r+bias
// ---------------------------------------------------------------------------
template <int MODE>
__global__ __launch_bounds__(128) void gemm_kernel(const _Float16* __restrict__ A,
                                                   const _Float16* __restrict__ BT,
                                                   const float* __restrict__ bias,
                                                   const float* __restrict__ residual,
                                                   float* __restrict__ outF,
                                                   _Float16* __restrict__ outH,
                                                   int M, int N, int K) {
  __shared__ __align__(16) _Float16 As[2][128 * 32];   // [m][k]
  __shared__ __align__(16) _Float16 Bs[2][128 * 32];   // [n][k]
  int t = threadIdx.x, lane = t & 31, wave = t >> 5;
  int m0 = blockIdx.x * 128, n0 = blockIdx.y * 128;
  int wm = (wave >> 1) * 64, wn = (wave & 1) * 64;
  int lo = lane & 15, hi = lane >> 4, kb = hi * 8;

  const _Float16* arow = A  + (size_t)(m0 + t) * K;   // one 32-half slab per thread
  const _Float16* brow = BT + (size_t)(n0 + t) * K;

  v8f acc[4][4];
#pragma unroll
  for (int i = 0; i < 4; ++i)
#pragma unroll
    for (int j = 0; j < 4; ++j)
#pragma unroll
      for (int r = 0; r < 8; ++r) acc[i][j][r] = 0.0f;

  // fill buffer 0 (k0 = 0)
#pragma unroll
  for (int j = 0; j < 4; ++j) {
    async_copy16(&As[0][t * 32 + j * 8], arow + j * 8);
    async_copy16(&Bs[0][t * 32 + j * 8], brow + j * 8);
  }

  int nk = K >> 5;
#pragma unroll 1
  for (int i = 0; i < nk; ++i) {
    int cur = i & 1;
    wait_async0();
    __syncthreads();                       // tile[cur] ready for everyone
    if (i + 1 < nk) {                      // overlap: fill tile[1-cur]
      int k1 = (i + 1) << 5;
#pragma unroll
      for (int j = 0; j < 4; ++j) {
        async_copy16(&As[1 - cur][t * 32 + j * 8], arow + k1 + j * 8);
        async_copy16(&Bs[1 - cur][t * 32 + j * 8], brow + k1 + j * 8);
      }
    }
    const _Float16* Ab = As[cur];
    const _Float16* Bb = Bs[cur];
    v16h a[4], b[4];
#pragma unroll
    for (int mt = 0; mt < 4; ++mt) {
      const _Float16* p = &Ab[(wm + mt * 16 + lo) * 32];
      a[mt] = load_a_pair(p + 2 * kb, p + 2 * kb + 8);
    }
#pragma unroll
    for (int nt = 0; nt < 4; ++nt)
      b[nt] = *(const v16h*)&Bb[(wn + nt * 16 + lo) * 32 + 16 * hi];
#pragma unroll
    for (int mt = 0; mt < 4; ++mt)
#pragma unroll
      for (int nt = 0; nt < 4; ++nt)
        acc[mt][nt] = wmma16(a[mt], b[nt], acc[mt][nt]);
  }

  // Epilogue. C layout: lane holds col N=lo, VGPR r holds row M=r+8*hi.
#pragma unroll
  for (int mt = 0; mt < 4; ++mt)
#pragma unroll
    for (int nt = 0; nt < 4; ++nt) {
      int n = n0 + wn + nt * 16 + lo;
      float bv = bias[n];
#pragma unroll
      for (int r = 0; r < 8; ++r) {
        int m = m0 + wm + mt * 16 + r + 8 * hi;
        float v = acc[mt][nt][r] + bv;
        size_t off = (size_t)m * N + n;
        if (MODE == 0) {
          outH[off] = (_Float16)v;
        } else if (MODE == 1) {
          outH[off] = (_Float16)(0.5f * v * (1.0f + erff(v * 0.70710678118f)));
        } else {
          outF[off] = residual[off] + v;
        }
      }
    }
}

// ---------------------------------------------------------------------------
// Flash-attention: block = 4 waves = 128 query rows of one (b,h); wave = 32 rows
// (2 M-tiles). Streams 32-key K/V blocks through LDS with online softmax.
// qkv layout: [token][ s*384 + h*64 + d ], s=0:q 1:k 2:v   (f16)
// ---------------------------------------------------------------------------
__global__ __launch_bounds__(128) void attn_kernel(const _Float16* __restrict__ qkv,
                                                   _Float16* __restrict__ attn_out) {
  __shared__ __align__(16) _Float16 Ks[32 * 64];        // [key][d]
  __shared__ __align__(16) _Float16 Vs[64 * 32];        // transposed: [d][key]
  __shared__ __align__(16) _Float16 Pb[4][2][16 * 32];  // per-wave P scratch

  int t = threadIdx.x, lane = t & 31, wave = t >> 5;
  int qblocks = SEQ / 128;                              // 16
  int bh = blockIdx.x / qblocks;
  int qblk = blockIdx.x - bh * qblocks;
  int b = bh / HEADS, h = bh % HEADS;
  size_t tokbase = (size_t)b * SEQ;
  int q0 = qblk * 128 + wave * 32;
  int lo = lane & 15, hi = lane >> 4, kb = hi * 8;

  // staging coords: thread t handles key row kk, d-range [db, db+16)
  int kk = t >> 2, db = (t & 3) * 16;

  // Q fragments: 2 M-tiles x 2 K-halves (Dh=64), pre-scaled by Dh^-0.5 = 2^-3
  v16h aq[2][2];
#pragma unroll
  for (int mt = 0; mt < 2; ++mt) {
    const _Float16* qrow = qkv + (tokbase + q0 + mt * 16 + lo) * CQKV + h * DH;
    aq[mt][0] = load_a_pair(qrow + 2 * kb,      qrow + 2 * kb + 8);
    aq[mt][1] = load_a_pair(qrow + 32 + 2 * kb, qrow + 40 + 2 * kb);
#pragma unroll
    for (int e = 0; e < 16; ++e) {
      aq[mt][0][e] = aq[mt][0][e] * (_Float16)0.125f;   // exact (power of 2)
      aq[mt][1][e] = aq[mt][1][e] * (_Float16)0.125f;
    }
  }

  v8f o[2][4];
  float mrow[2][8], lrow[2][8];
#pragma unroll
  for (int mt = 0; mt < 2; ++mt) {
#pragma unroll
    for (int r = 0; r < 8; ++r) { mrow[mt][r] = -3.0e38f; lrow[mt][r] = 0.0f; }
#pragma unroll
    for (int dt = 0; dt < 4; ++dt)
#pragma unroll
      for (int r = 0; r < 8; ++r) o[mt][dt][r] = 0.0f;
  }

  for (int k0 = 0; k0 < SEQ; k0 += 32) {
    __syncthreads();                       // previous block fully consumed
    {
      // K block via async global->LDS (16B per lane)
      const _Float16* krow = qkv + (tokbase + k0 + kk) * CQKV + DIMC + h * DH + db;
      async_copy16(&Ks[kk * 64 + db],     krow);
      async_copy16(&Ks[kk * 64 + db + 8], krow + 8);
      // V block: manual transpose into [d][key]
      const _Float16* vrow = qkv + (tokbase + k0 + kk) * CQKV + 2 * DIMC + h * DH + db;
      v8h vv0 = *(const v8h*)vrow;
      v8h vv1 = *(const v8h*)(vrow + 8);
#pragma unroll
      for (int j = 0; j < 8; ++j) {
        Vs[(db + j) * 32 + kk]     = vv0[j];
        Vs[(db + 8 + j) * 32 + kk] = vv1[j];
      }
    }
    wait_async0();
    __syncthreads();

    // S = Q*K^T : 2 M-tiles x 2 key tiles, K-dim 64 = 2 WMMAs each (8 total)
    v16h bk[2][2];
#pragma unroll
    for (int kt = 0; kt < 2; ++kt) {
      bk[kt][0] = *(const v16h*)&Ks[(kt * 16 + lo) * 64 + 16 * hi];
      bk[kt][1] = *(const v16h*)&Ks[(kt * 16 + lo) * 64 + 32 + 16 * hi];
    }
    v8f s[2][2];
#pragma unroll
    for (int mt = 0; mt < 2; ++mt)
#pragma unroll
      for (int kt = 0; kt < 2; ++kt) {
#pragma unroll
        for (int r = 0; r < 8; ++r) s[mt][kt][r] = 0.0f;
        s[mt][kt] = wmma16(aq[mt][0], bk[kt][0], s[mt][kt]);
        s[mt][kt] = wmma16(aq[mt][1], bk[kt][1], s[mt][kt]);
      }

    // online softmax per row (rows live in 16-lane groups)
#pragma unroll
    for (int mt = 0; mt < 2; ++mt)
#pragma unroll
      for (int r = 0; r < 8; ++r) {
        float x0 = s[mt][0][r], x1 = s[mt][1][r];
        float mx = fmaxf(x0, x1);
#pragma unroll
        for (int msk = 1; msk < 16; msk <<= 1) mx = fmaxf(mx, __shfl_xor(mx, msk, 32));
        float mnew = fmaxf(mrow[mt][r], mx);
        float p0 = __expf(x0 - mnew), p1 = __expf(x1 - mnew);
        float ps = p0 + p1;
#pragma unroll
        for (int msk = 1; msk < 16; msk <<= 1) ps += __shfl_xor(ps, msk, 32);
        float alpha = __expf(mrow[mt][r] - mnew);
        lrow[mt][r] = lrow[mt][r] * alpha + ps;
        mrow[mt][r] = mnew;
#pragma unroll
        for (int dt = 0; dt < 4; ++dt) o[mt][dt][r] *= alpha;
        Pb[wave][mt][(r + 8 * hi) * 32 + lo]      = (_Float16)p0;
        Pb[wave][mt][(r + 8 * hi) * 32 + 16 + lo] = (_Float16)p1;
      }

    // O += P*V : 2 M-tiles x 4 d-tiles (8 WMMAs), K-dim = 32 keys
    v16h bv[4];
#pragma unroll
    for (int dt = 0; dt < 4; ++dt)
      bv[dt] = *(const v16h*)&Vs[(dt * 16 + lo) * 32 + 16 * hi];
#pragma unroll
    for (int mt = 0; mt < 2; ++mt) {
      const _Float16* pr = &Pb[wave][mt][lo * 32];
      v16h ap = load_a_pair(pr + 2 * kb, pr + 2 * kb + 8);
#pragma unroll
      for (int dt = 0; dt < 4; ++dt)
        o[mt][dt] = wmma16(ap, bv[dt], o[mt][dt]);
    }
  }

  // normalize + store f16 [token][h*64+d]
#pragma unroll
  for (int mt = 0; mt < 2; ++mt)
#pragma unroll
    for (int r = 0; r < 8; ++r) {
      float inv = 1.0f / lrow[mt][r];
      size_t row = (tokbase + q0 + mt * 16 + r + 8 * hi) * (size_t)DIMC + h * DH;
#pragma unroll
      for (int dt = 0; dt < 4; ++dt)
        attn_out[row + dt * 16 + lo] = (_Float16)(o[mt][dt][r] * inv);
    }
}

// ---------------------------------------------------------------------------
// Launcher
// ---------------------------------------------------------------------------
extern "C" void kernel_launch(void* const* d_in, const int* in_sizes, int n_in,
                              void* d_out, int out_size, void* d_ws, size_t ws_size,
                              hipStream_t stream) {
  (void)in_sizes; (void)n_in; (void)out_size; (void)ws_size;
  const float* x      = (const float*)d_in[0];
  const float* w_qkv  = (const float*)d_in[1];
  const float* b_qkv  = (const float*)d_in[2];
  const float* w_proj = (const float*)d_in[3];
  const float* b_proj = (const float*)d_in[4];
  const float* w_fc1  = (const float*)d_in[5];
  const float* b_fc1  = (const float*)d_in[6];
  const float* w_fc2  = (const float*)d_in[7];
  const float* b_fc2  = (const float*)d_in[8];
  const float* g1     = (const float*)d_in[9];
  const float* be1    = (const float*)d_in[10];
  const float* g2     = (const float*)d_in[11];
  const float* be2    = (const float*)d_in[12];
  float* out = (float*)d_out;

  char* ws = (char*)d_ws;
  size_t off = 0;
  auto alloc = [&](size_t bytes) -> void* {
    void* p = ws + off;
    off = (off + bytes + 255) & ~(size_t)255;
    return p;
  };
  _Float16* wqh  = (_Float16*)alloc((size_t)DIMC * CQKV * 2);  // [N][K]
  _Float16* wph  = (_Float16*)alloc((size_t)DIMC * DIMC * 2);
  _Float16* w1h  = (_Float16*)alloc((size_t)DIMC * HID  * 2);
  _Float16* w2h  = (_Float16*)alloc((size_t)HID  * DIMC * 2);
  _Float16* xn   = (_Float16*)alloc((size_t)MTOT * DIMC * 2);
  _Float16* qkv  = (_Float16*)alloc((size_t)MTOT * CQKV * 2);
  _Float16* attn = (_Float16*)alloc((size_t)MTOT * DIMC * 2);
  float*    res1 = (float*)   alloc((size_t)MTOT * DIMC * 4);
  _Float16* xn2  = (_Float16*)alloc((size_t)MTOT * DIMC * 2);
  _Float16* h1   = (_Float16*)alloc((size_t)MTOT * HID  * 2);

  cvt_t_kernel<<<(DIMC * CQKV + 255) / 256, 256, 0, stream>>>(w_qkv,  wqh, DIMC, CQKV);
  cvt_t_kernel<<<(DIMC * DIMC + 255) / 256, 256, 0, stream>>>(w_proj, wph, DIMC, DIMC);
  cvt_t_kernel<<<(DIMC * HID  + 255) / 256, 256, 0, stream>>>(w_fc1,  w1h, DIMC, HID);
  cvt_t_kernel<<<(HID  * DIMC + 255) / 256, 256, 0, stream>>>(w_fc2,  w2h, HID, DIMC);

  // x -> LN1 -> xn (f16)
  ln_kernel<<<MTOT, 128, 0, stream>>>(x, g1, be1, xn);
  // qkv = xn @ w_qkv + b_qkv  (f16 out)
  gemm_kernel<0><<<dim3(MTOT / 128, CQKV / 128), 128, 0, stream>>>(
      xn, wqh, b_qkv, nullptr, nullptr, qkv, MTOT, CQKV, DIMC);
  // attention
  attn_kernel<<<BATCH * HEADS * (SEQ / 128), 128, 0, stream>>>(qkv, attn);
  // res1 = x + attn @ w_proj + b_proj  (f32)
  gemm_kernel<2><<<dim3(MTOT / 128, DIMC / 128), 128, 0, stream>>>(
      attn, wph, b_proj, x, res1, nullptr, MTOT, DIMC, DIMC);
  // LN2
  ln_kernel<<<MTOT, 128, 0, stream>>>(res1, g2, be2, xn2);
  // h1 = gelu(xn2 @ w_fc1 + b_fc1)  (f16)
  gemm_kernel<1><<<dim3(MTOT / 128, HID / 128), 128, 0, stream>>>(
      xn2, w1h, b_fc1, nullptr, nullptr, h1, MTOT, HID, DIMC);
  // out = res1 + h1 @ w_fc2 + b_fc2  (f32)
  gemm_kernel<2><<<dim3(MTOT / 128, DIMC / 128), 128, 0, stream>>>(
      h1, w2h, b_fc2, res1, out, nullptr, MTOT, DIMC, HID);
}
